// EvolutionarySelector_69277822485300
// MI455X (gfx1250) — matrix operand order repaired and checked by reading, not compiled, not run on hardware
//
#include <hip/hip_runtime.h>
#include <hip/hip_bf16.h>
#include <stdint.h>

typedef __attribute__((ext_vector_type(16))) __bf16 v16bf;
typedef __attribute__((ext_vector_type(4)))  __bf16 v4bf;
typedef __attribute__((ext_vector_type(4)))  float  v4f;
typedef __attribute__((ext_vector_type(8)))  float  v8f;

#define NQ 2048
#define NM 65536
#define DIM 1024
#define TOPK 5
#define MT 128                    // q rows per workgroup
#define NT 128                    // bank rows per sub-tile
#define KT 32                     // K per WMMA stage step
#define KPAD 40                   // LDS row stride in bf16 elems (80B: 16B-aligned, low-conflict)
#define SIM_LD 66                 // padded sim row stride (floats), 64 cols per pass
#define NCHUNK 32
#define CHUNK (NM / NCHUNK)       // 2048 bank rows per workgroup
#define QBLOCKS (NQ / MT)         // 16
#define NSUB (CHUNK / NT)         // 16

// ---------------- helpers ----------------

__device__ __forceinline__ void topk_insert(float val, int idx, float (&tv)[TOPK], int (&ti)[TOPK]) {
  if (val <= tv[4]) return;
  if (val > tv[0]) {
    tv[4]=tv[3]; ti[4]=ti[3]; tv[3]=tv[2]; ti[3]=ti[2];
    tv[2]=tv[1]; ti[2]=ti[1]; tv[1]=tv[0]; ti[1]=ti[0];
    tv[0]=val;  ti[0]=idx;
  } else if (val > tv[1]) {
    tv[4]=tv[3]; ti[4]=ti[3]; tv[3]=tv[2]; ti[3]=ti[2];
    tv[2]=tv[1]; ti[2]=ti[1]; tv[1]=val;  ti[1]=idx;
  } else if (val > tv[2]) {
    tv[4]=tv[3]; ti[4]=ti[3]; tv[3]=tv[2]; ti[3]=ti[2];
    tv[2]=val;  ti[2]=idx;
  } else if (val > tv[3]) {
    tv[4]=tv[3]; ti[4]=ti[3]; tv[3]=val; ti[3]=idx;
  } else {
    tv[4]=val;  ti[4]=idx;
  }
}

__device__ __forceinline__ uint32_t pcg_hash(uint32_t x) {
  x = x * 747796405u + 2891336453u;
  uint32_t w = ((x >> ((x >> 28u) + 4u)) ^ x) * 277803737u;
  return (w >> 22u) ^ w;
}

// packed f32x4 -> bf16x4 via vector fptrunc (targets v_cvt_pk_bf16_f32 pairs)
__device__ __forceinline__ uint2 cvt4_bf16(float4 f) {
  v4f vf; vf[0] = f.x; vf[1] = f.y; vf[2] = f.z; vf[3] = f.w;
  union { v4bf b; uint2 u; } un;
  un.b = __builtin_convertvector(vf, v4bf);
  return un.u;
}

union Frag { v16bf v; uint4 q[2]; };

// ---------------- kernel 1: inverse L2 norms ----------------

__global__ void norms_kernel(const float* __restrict__ q, const float* __restrict__ m,
                             float* __restrict__ inv_q, float* __restrict__ inv_m) {
  __shared__ float red[8];
  const int row = blockIdx.x;
  const float* src = (row < NQ) ? (q + (size_t)row * DIM) : (m + (size_t)(row - NQ) * DIM);
  const float4* s4 = (const float4*)src;
  float4 v = s4[threadIdx.x];
  float ss = v.x*v.x + v.y*v.y + v.z*v.z + v.w*v.w;
  #pragma unroll
  for (int o = 16; o > 0; o >>= 1) ss += __shfl_down(ss, o, 32);
  if ((threadIdx.x & 31) == 0) red[threadIdx.x >> 5] = ss;
  __syncthreads();
  if (threadIdx.x == 0) {
    float t = 0.f;
    #pragma unroll
    for (int i = 0; i < 8; ++i) t += red[i];
    float inv = 1.0f / fmaxf(sqrtf(t), 1e-8f);
    if (row < NQ) inv_q[row] = inv; else inv_m[row - NQ] = inv;
  }
}

// ---------------- kernel 2: bf16 WMMA similarity + streaming top-5 ----------------

__global__ void __launch_bounds__(256)
simtopk_kernel(const float* __restrict__ q, const float* __restrict__ m,
               const float* __restrict__ inv_q, const float* __restrict__ inv_m,
               float* __restrict__ cand_val, int* __restrict__ cand_idx) {
  __shared__ unsigned short Abuf[MT * KPAD];   // 10240 B : 128 rows x 32K bf16 (padded)
  __shared__ unsigned short Bbuf[NT * KPAD];   // 10240 B
  __shared__ float simb[MT * SIM_LD];          // 33792 B : 128 rows x 64 cols (padded)
  __shared__ float cval[MT * 2 * TOPK];        // 5120 B
  __shared__ int   cidx[MT * 2 * TOPK];        // 5120 B   -> total 64512 B

  const int tid  = threadIdx.x;
  const int wave = tid >> 5, lane = tid & 31;
  const int lh   = lane >> 4;                  // K half selector
  const int lmod = lane & 15;                  // fragment row/col within tile
  const int wm   = wave >> 1;                  // 0..3 : 32-row group
  const int wn   = wave & 1;                   // 0..1 : 64-col group

  const int qb      = blockIdx.x % QBLOCKS;    // q-block fastest -> chunk shared in L2
  const int chunk   = blockIdx.x / QBLOCKS;
  const int qbase   = qb * MT;
  const int nchunk0 = chunk * CHUNK;

  // staging / scan thread mapping: (row, half)
  const int srow = tid >> 1;                   // 0..127
  const int sh   = tid & 1;                    // K/col half

  const float* aptr = q + (size_t)(qbase + srow) * DIM + 16 * sh;

  // inv_q for the 16 accumulator rows this lane owns: row = wm*32 + r*16 + 8*lh + v
  float iqv[2][8];
  #pragma unroll
  for (int r = 0; r < 2; ++r)
    #pragma unroll
    for (int v = 0; v < 8; ++v)
      iqv[r][v] = inv_q[qbase + wm * 32 + r * 16 + 8 * lh + v];

  float tv[TOPK]; int ti[TOPK];
  #pragma unroll
  for (int i = 0; i < TOPK; ++i) { tv[i] = -3.0e38f; ti[i] = 0; }

  #pragma unroll 1
  for (int ns = 0; ns < NSUB; ++ns) {
    const int nbase = nchunk0 + ns * NT;
    const float* bptr = m + (size_t)(nbase + srow) * DIM + 16 * sh;

    v8f acc[2][4];
    #pragma unroll
    for (int r = 0; r < 2; ++r)
      #pragma unroll
      for (int t = 0; t < 4; ++t)
        acc[r][t] = (v8f){0.f,0.f,0.f,0.f,0.f,0.f,0.f,0.f};

    // software pipeline: preload K-step 0 into registers
    float4 pa[4], pb[4];
    {
      const float4* asrc = (const float4*)(aptr);
      const float4* bsrc = (const float4*)(bptr);
      #pragma unroll
      for (int i = 0; i < 4; ++i) { pa[i] = asrc[i]; pb[i] = bsrc[i]; }
    }

    #pragma unroll 1
    for (int k0 = 0; k0 < DIM; k0 += KT) {
      __syncthreads();  // previous fragment reads done before restaging
      {  // convert + store staged K-step
        uint2* da = (uint2*)&Abuf[srow * KPAD + 16 * sh];
        uint2* db = (uint2*)&Bbuf[srow * KPAD + 16 * sh];
        #pragma unroll
        for (int i = 0; i < 4; ++i) { da[i] = cvt4_bf16(pa[i]); db[i] = cvt4_bf16(pb[i]); }
      }
      __syncthreads();

      // prefetch next K-step while WMMAs run
      if (k0 + KT < DIM) {
        const float4* asrc = (const float4*)(aptr + k0 + KT);
        const float4* bsrc = (const float4*)(bptr + k0 + KT);
        #pragma unroll
        for (int i = 0; i < 4; ++i) { pa[i] = asrc[i]; pb[i] = bsrc[i]; }
      }

      // A fragments for the two 16-row tiles (16-bit A layout: runs at K=8lh and K=16+8lh)
      Frag a0, a1;
      const unsigned short* ar0 = &Abuf[(wm * 32 + lmod) * KPAD];
      const unsigned short* ar1 = &Abuf[(wm * 32 + 16 + lmod) * KPAD];
      a0.q[0] = *(const uint4*)(ar0 + 8 * lh);
      a0.q[1] = *(const uint4*)(ar0 + 16 + 8 * lh);
      a1.q[0] = *(const uint4*)(ar1 + 8 * lh);
      a1.q[1] = *(const uint4*)(ar1 + 16 + 8 * lh);

      #pragma unroll
      for (int t = 0; t < 4; ++t) {
        // B fragment (B layout: runs at K=8lh and K=16+8lh for col n)
        Frag b;
        const unsigned short* br = &Bbuf[(wn * 64 + t * 16 + lmod) * KPAD];
        b.q[0] = *(const uint4*)(br + 8 * lh);
        b.q[1] = *(const uint4*)(br + 16 + 8 * lh);
        acc[0][t] = __builtin_amdgcn_wmma_f32_16x16x32_bf16(
            false, a0.v, false, b.v, (short)0, acc[0][t], false, false);
        acc[1][t] = __builtin_amdgcn_wmma_f32_16x16x32_bf16(
            false, a1.v, false, b.v, (short)0, acc[1][t], false, false);
      }
    }

    // two column passes: scale to cosine, stage 128x64 in LDS, streaming top-5
    #pragma unroll 1
    for (int p = 0; p < 2; ++p) {
      __syncthreads();
      if (wn == p) {
        #pragma unroll
        for (int t = 0; t < 4; ++t) {
          const int c = t * 16 + lmod;
          const float im = inv_m[nbase + p * 64 + c];
          #pragma unroll
          for (int r = 0; r < 2; ++r)
            #pragma unroll
            for (int v = 0; v < 8; ++v) {
              const int rl = wm * 32 + r * 16 + 8 * lh + v;
              simb[rl * SIM_LD + c] = acc[r][t][v] * iqv[r][v] * im;
            }
        }
      }
      __syncthreads();
      // scan 32 cols per thread as float2 (8B-aligned: SIM_LD even)
      const float2* srow2 = (const float2*)&simb[srow * SIM_LD + sh * 32];
      const int cbase = nbase + p * 64 + sh * 32;
      #pragma unroll 1
      for (int j = 0; j < 16; ++j) {
        float2 s2 = srow2[j];
        topk_insert(s2.x, cbase + 2 * j,     tv, ti);
        topk_insert(s2.y, cbase + 2 * j + 1, tv, ti);
      }
    }
  }

  // merge the 2 partial top-5s per row -> per-(row,chunk) candidates
  #pragma unroll
  for (int i = 0; i < TOPK; ++i) { cval[tid * TOPK + i] = tv[i]; cidx[tid * TOPK + i] = ti[i]; }
  __syncthreads();
  if (tid < MT) {
    float fv[TOPK]; int fi[TOPK];
    #pragma unroll
    for (int i = 0; i < TOPK; ++i) { fv[i] = -3.0e38f; fi[i] = 0; }
    #pragma unroll 1
    for (int e = 0; e < 2 * TOPK; ++e)
      topk_insert(cval[tid * 2 * TOPK + e], cidx[tid * 2 * TOPK + e], fv, fi);
    const size_t out = ((size_t)(qbase + tid) * NCHUNK + chunk) * TOPK;
    #pragma unroll
    for (int i = 0; i < TOPK; ++i) { cand_val[out + i] = fv[i]; cand_idx[out + i] = fi[i]; }
  }
}

// ---------------- kernel 3: merge chunk candidates -> final top-5 ----------------

__global__ void merge_kernel(const float* __restrict__ cand_val, const int* __restrict__ cand_idx,
                             int* __restrict__ top_idx) {
  const int row = blockIdx.x * blockDim.x + threadIdx.x;
  if (row >= NQ) return;
  const float* cv = cand_val + (size_t)row * NCHUNK * TOPK;
  const int*   ci = cand_idx + (size_t)row * NCHUNK * TOPK;
  float fv[TOPK]; int fi[TOPK];
  #pragma unroll
  for (int i = 0; i < TOPK; ++i) { fv[i] = -3.0e38f; fi[i] = 0; }
  #pragma unroll 1
  for (int e = 0; e < NCHUNK * TOPK; ++e) topk_insert(cv[e], ci[e], fv, fi);
  #pragma unroll
  for (int i = 0; i < TOPK; ++i) top_idx[row * TOPK + i] = fi[i];
}

// ---------------- kernel 4: gather + deterministic mutation ----------------

__global__ void gather_mutate_kernel(const float* __restrict__ m, const int* __restrict__ top_idx,
                                     float* __restrict__ out) {
  const int rk  = blockIdx.x;               // 0 .. NQ*TOPK-1
  const int idx = top_idx[rk];
  const float4* src = (const float4*)(m + (size_t)idx * DIM);
  float4 v = src[threadIdx.x];
  const size_t obase = (size_t)rk * DIM + (size_t)threadIdx.x * 4;
  float vals[4] = {v.x, v.y, v.z, v.w};
  #pragma unroll
  for (int c = 0; c < 4; ++c) {
    const uint32_t e  = (uint32_t)(obase + c);
    const float um = (pcg_hash(e ^ 0x9E3779B9u) >> 8) * (1.0f / 16777216.0f);
    if (um < 0.1f) {  // MUTATION_RATE
      float u1 = fmaxf((pcg_hash(e ^ 0x85EBCA6Bu) >> 8) * (1.0f / 16777216.0f), 1e-7f);
      float u2 = (pcg_hash(e ^ 0xC2B2AE35u) >> 8) * (1.0f / 16777216.0f);
      float nrm = sqrtf(-2.0f * logf(u1)) * cosf(6.28318530718f * u2);
      vals[c] += nrm * 0.05f;  // NOISE_SCALE
    }
  }
  float4 o; o.x = vals[0]; o.y = vals[1]; o.z = vals[2]; o.w = vals[3];
  ((float4*)(out + (size_t)rk * DIM))[threadIdx.x] = o;
}

// ---------------- launch ----------------

extern "C" void kernel_launch(void* const* d_in, const int* in_sizes, int n_in,
                              void* d_out, int out_size, void* d_ws, size_t ws_size,
                              hipStream_t stream) {
  const float* q = (const float*)d_in[0];   // current_feat [2048,1024]
  const float* m = (const float*)d_in[1];   // memory_bank  [65536,1024]
  float* out = (float*)d_out;               // [2048,5,1024]

  char* ws = (char*)d_ws;
  float* inv_q    = (float*)(ws);                                   // 2048 f
  float* inv_m    = (float*)(ws + 8192);                            // 65536 f
  float* cand_val = (float*)(ws + 270336);                          // 2048*32*5 f
  int*   cand_idx = (int*)  (ws + 270336 + 1310720);                // 2048*32*5 i
  int*   top_idx  = (int*)  (ws + 270336 + 2 * 1310720);            // 2048*5 i

  norms_kernel<<<NQ + NM, 256, 0, stream>>>(q, m, inv_q, inv_m);
  simtopk_kernel<<<QBLOCKS * NCHUNK, 256, 0, stream>>>(q, m, inv_q, inv_m, cand_val, cand_idx);
  merge_kernel<<<(NQ + 255) / 256, 256, 0, stream>>>(cand_val, cand_idx, top_idx);
  gather_mutate_kernel<<<NQ * TOPK, 256, 0, stream>>>(m, top_idx, out);
}
